// FourierDecoder_15006615732484
// MI455X (gfx1250) — matrix operand order repaired and checked by reading, not compiled, not run on hardware
//
#include <hip/hip_runtime.h>
#include <hip/hip_bf16.h>

typedef __attribute__((ext_vector_type(16))) _Float16 v16h;
typedef __attribute__((ext_vector_type(8)))  _Float16 v8h;
typedef __attribute__((ext_vector_type(8)))  float    v8f;

#define BATCH   8192
#define LATENT  64
#define HIDDEN  64
#define NCOEF   128           // N_FREQ*N_FREQ*2
#define IMG     64
#define NPIX    (IMG*IMG)     // 4096
#define LDS_STRIDE 136        // halves per basis column in LDS (272B: bank-conflict-free)

// ---------------------------------------------------------------------------
// Kernel B: Fourier basis, TRANSPOSED: basisT[n][k], n in [0,4096), k in [0,128)
// k = 16*fx + 2*fy + ch ; ch==0 -> cos, ch==1 -> sin ; phase = pi/32*(fx*x+fy*y)
// ---------------------------------------------------------------------------
__global__ __launch_bounds__(256) void build_basis(_Float16* __restrict__ basisT)
{
    const int idx = blockIdx.x * 256 + threadIdx.x;   // 0 .. 4096*128-1
    const int n  = idx >> 7;
    const int k  = idx & 127;
    const int f  = k >> 1;
    const int fx = f >> 3;
    const int fy = f & 7;
    const int x  = n >> 6;
    const int y  = n & 63;
    const float phase = 0.09817477042f * (float)(fx * x + fy * y);  // pi/32
    const float v = (k & 1) ? sinf(phase) : cosf(phase);
    basisT[idx] = (_Float16)v;
}

// ---------------------------------------------------------------------------
// Kernel A: coeffs[b][c] = (relu(z@W1+b1) @ W2 + b2), stored f16.
// ---------------------------------------------------------------------------
__global__ __launch_bounds__(256) void mlp_head(
    const float* __restrict__ z,    // [8192][64]
    const float* __restrict__ W1,   // [64][64]
    const float* __restrict__ b1,   // [64]
    const float* __restrict__ W2,   // [64][128]
    const float* __restrict__ b2,   // [128]
    _Float16*    __restrict__ coeffs) // [8192][128] f16
{
    __shared__ float zsh[4][LATENT];
    __shared__ float hsh[4][HIDDEN];
    const int lr  = threadIdx.x >> 6;        // local row 0..3
    const int j   = threadIdx.x & 63;
    const int row = (blockIdx.x << 2) + lr;

    zsh[lr][j] = z[row * LATENT + j];
    __syncthreads();

    float h = b1[j];
#pragma unroll
    for (int k = 0; k < LATENT; ++k)
        h = fmaf(zsh[lr][k], W1[k * HIDDEN + j], h);
    hsh[lr][j] = fmaxf(h, 0.0f);
    __syncthreads();

    float c0 = b2[j];
    float c1 = b2[j + 64];
#pragma unroll
    for (int k = 0; k < HIDDEN; ++k) {
        const float hv = hsh[lr][k];
        c0 = fmaf(hv, W2[k * NCOEF + j],      c0);
        c1 = fmaf(hv, W2[k * NCOEF + j + 64], c1);
    }
    coeffs[row * NCOEF + j]      = (_Float16)c0;
    coeffs[row * NCOEF + j + 64] = (_Float16)c1;
}

// ---------------------------------------------------------------------------
// Kernel C: out = sigmoid( (coeffs @ basisT^T) / 8 )   [8192 x 4096, K=128]
//
// Block = 8 waves = 128(M) x 64(N) tile; all waves share the same 64 basis
// columns. The 16KB basis tile is staged into LDS ONCE per block with the
// CDNA5 async-to-LDS path (ASYNCcnt), then each wave runs 16 WMMAs with
// B fragments served from LDS (row stride padded to 272B => conflict-free).
// ---------------------------------------------------------------------------
__global__ __launch_bounds__(256) void fourier_gemm(
    const _Float16* __restrict__ coeffs,   // [8192][128] f16
    const _Float16* __restrict__ basisT,   // [4096][128] f16
    float*          __restrict__ out)      // [8192][4096] f32
{
    __shared__ _Float16 btile[64 * LDS_STRIDE];   // 17408 B

    const int tid  = threadIdx.x;
    const int lane = tid & 31;
    const int half = lane >> 4;             // 0 | 1
    const int l16  = lane & 15;
    const int wid  = tid >> 5;              // wave in block: 0..7

    const int bn = blockIdx.x & 63;         // N block: 64 cols
    const int bm = blockIdx.x >> 6;         // M block: 128 rows
    const int n0 = bn << 6;
    const int m0 = (bm << 7) + (wid << 4);  // this wave's 16-row tile

    // ---- async copy: 64 basis columns (128 halves each) global -> LDS ----
    // thread t: column c = t>>2, chunk i = 4*(t&3) + it, 16B per chunk
    {
        const int c    = tid >> 2;
        const int part = tid & 3;
        const _Float16* gcol = basisT + (n0 + c) * NCOEF;
#pragma unroll
        for (int it = 0; it < 4; ++it) {
            const int i = part * 4 + it;                 // chunk 0..15
            const _Float16* gsrc = gcol + i * 8;         // 8 halves = 16B
            const unsigned   loff =
                (unsigned)(size_t)(&btile[c * LDS_STRIDE + i * 8]);
            asm volatile("global_load_async_to_lds_b128 %0, %1, off"
                         :: "v"(loff), "v"(gsrc) : "memory");
        }
    }

    // ---- A fragments while the copy is in flight: row m0+l16, 4 K-chunks ----
    const _Float16* arow = coeffs + (m0 + l16) * NCOEF;
    v16h a[4];
#pragma unroll
    for (int kk = 0; kk < 4; ++kk) {
        const int kc = kk * 32;
        const v8h lo = *(const v8h*)(arow + kc + half * 8);
        const v8h hi = *(const v8h*)(arow + kc + 16 + half * 8);
        v16h f;
#pragma unroll
        for (int i = 0; i < 8; ++i) { f[i] = lo[i]; f[i + 8] = hi[i]; }
        a[kk] = f;
    }

    // drain async copies, then block-wide barrier before consuming LDS
    asm volatile("s_wait_asynccnt 0x0" ::: "memory");
    __syncthreads();

#pragma unroll
    for (int s = 0; s < 4; ++s) {
        const int n0s = n0 + s * 16;
        // B fragment base in LDS: col (s*16+l16), K offset half*8
        const _Float16* bcol =
            &btile[(s * 16 + l16) * LDS_STRIDE + half * 8];

        v8f acc = {};
#pragma unroll
        for (int kk = 0; kk < 4; ++kk) {
            const v8h blo = *(const v8h*)(bcol + kk * 32);
            const v8h bhi = *(const v8h*)(bcol + kk * 32 + 16);
            v16h b;
#pragma unroll
            for (int i = 0; i < 8; ++i) { b[i] = blo[i]; b[i + 8] = bhi[i]; }
            acc = __builtin_amdgcn_wmma_f32_16x16x32_f16(
                      /*neg_a=*/false, a[kk],
                      /*neg_b=*/false, b,
                      /*c_mod=*/(short)0, acc,
                      /*reuse_a=*/false, /*reuse_b=*/false);
        }

        // sigmoid(acc/8); non-temporal streaming stores (write-once 128MB)
        float* orow = out + (m0 + half * 8) * NPIX + n0s + l16;
#pragma unroll
        for (int v = 0; v < 8; ++v) {
            const float x  = acc[v] * 0.125f;
            const float sg = 1.0f / (1.0f + __expf(-x));
            __builtin_nontemporal_store(sg, orow + v * NPIX);
        }
    }
}

// ---------------------------------------------------------------------------
extern "C" void kernel_launch(void* const* d_in, const int* in_sizes, int n_in,
                              void* d_out, int out_size, void* d_ws, size_t ws_size,
                              hipStream_t stream)
{
    const float* z  = (const float*)d_in[0];
    const float* W1 = (const float*)d_in[1];
    const float* b1 = (const float*)d_in[2];
    const float* W2 = (const float*)d_in[3];
    const float* b2 = (const float*)d_in[4];
    float* out = (float*)d_out;

    _Float16* basisT = (_Float16*)d_ws;                                     // 1 MB
    _Float16* coeffs = (_Float16*)((char*)d_ws + (size_t)NPIX * NCOEF * 2); // 2 MB

    build_basis<<<(NPIX * NCOEF) / 256, 256, 0, stream>>>(basisT);
    mlp_head<<<BATCH / 4, 256, 0, stream>>>(z, W1, b1, W2, b2, coeffs);
    // 64 M-blocks * 64 N-blocks
    fourier_gemm<<<64 * 64, 256, 0, stream>>>(coeffs, basisT, out);
}
